// GroupedQueryAttention_71425306133068
// MI455X (gfx1250) — matrix operand order repaired and checked by reading, not compiled, not run on hardware
//
#include <hip/hip_runtime.h>
#include <hip/hip_bf16.h>

// ---------------------------------------------------------------------------
// GQA attention layer for MI455X (gfx1250), wave32 + WMMA bf16.
// B=1, S=4096, D_MODEL=2048, 16 Q heads / 4 KV heads, HEAD_DIM=128.
// ---------------------------------------------------------------------------

#define S_LEN   4096
#define DMODEL  2048
#define NHEADS  16
#define NKV     4
#define HDIM    128

typedef unsigned short u16;
typedef __bf16 v8bf  __attribute__((ext_vector_type(8)));
typedef __bf16 v16bf __attribute__((ext_vector_type(16)));
typedef float  v8f   __attribute__((ext_vector_type(8)));

static __device__ __forceinline__ u16 f2bf_bits(float f) {
    unsigned u = __builtin_bit_cast(unsigned, f);
    u += 0x7FFFu + ((u >> 16) & 1u);          // round-to-nearest-even
    return (u16)(u >> 16);
}
static __device__ __forceinline__ __bf16 f2bf(float f) {
    u16 b = f2bf_bits(f);
    return __builtin_bit_cast(__bf16, b);
}
static __device__ __forceinline__ float bf2f(u16 b) {
    unsigned u = ((unsigned)b) << 16;
    return __builtin_bit_cast(float, u);
}

// A-matrix fragment (16x32 bf16, M=lane%16). Per ISA 7.12.2:
// lane group g = lane/16; element e -> K = (e<8 ? e : e+8) + g*8
// => two contiguous 8-element (16B) loads at K = g*8 and K = 16+g*8.
static __device__ __forceinline__ v16bf ld_frag_row(const u16* p, int g) {
    v8bf a = *(const v8bf*)(p + g * 8);
    v8bf b = *(const v8bf*)(p + 16 + g * 8);
    return __builtin_shufflevector(a, b, 0,1,2,3,4,5,6,7,8,9,10,11,12,13,14,15);
}
// B-matrix fragment (32x16 bf16, N=lane%16):
// lane group g holds K = (e<8 ? e : e+8) + g*8 (same order as the A fragment).
static __device__ __forceinline__ v16bf ld_frag_col(const u16* p, int g) {
    v8bf a = *(const v8bf*)(p + g * 8);
    v8bf b = *(const v8bf*)(p + 16 + g * 8);
    return __builtin_shufflevector(a, b, 0,1,2,3,4,5,6,7,8,9,10,11,12,13,14,15);
}
static __device__ __forceinline__ v8f wmma_bf16(v16bf a, v16bf b, v8f c) {
    return __builtin_amdgcn_wmma_f32_16x16x32_bf16(false, a, false, b, (short)0, c,
                                                   false, false);
}

// ---------------------------------------------------------------------------
// Elementwise prep kernels
// ---------------------------------------------------------------------------
__global__ void f32_to_bf16_k(const float* __restrict__ src, u16* __restrict__ dst, int n) {
    int i = blockIdx.x * blockDim.x + threadIdx.x;
    if (i < n) dst[i] = f2bf_bits(src[i]);
}

// src is [K][N] row-major fp32; dst is [N][K] bf16 (transposed for B-frag loads)
__global__ void f32_transpose_bf16_k(const float* __restrict__ src, u16* __restrict__ dst,
                                     int K, int N) {
    int i = blockIdx.x * blockDim.x + threadIdx.x;
    if (i >= K * N) return;
    int k = i / N, n = i - k * N;
    dst[(size_t)n * K + k] = f2bf_bits(src[i]);
}

// RoPE + relayout: src [S][Hn*128] bf16 -> dst [Hn][S][128] bf16
__global__ void rope_relayout_k(const u16* __restrict__ src, const float* __restrict__ cosT,
                                const float* __restrict__ sinT, u16* __restrict__ dst, int Hn) {
    int tid = blockIdx.x * blockDim.x + threadIdx.x;
    if (tid >= S_LEN * Hn * 64) return;
    int j = tid & 63;
    int h = (tid >> 6) % Hn;
    int s = tid / (Hn * 64);
    const u16* sp = src + (size_t)s * (Hn * 128) + h * 128;
    float x0 = bf2f(sp[j]), x1 = bf2f(sp[j + 64]);
    float c0 = cosT[s * 128 + j], c1 = cosT[s * 128 + j + 64];
    float s0 = sinT[s * 128 + j], s1 = sinT[s * 128 + j + 64];
    u16* dp = dst + ((size_t)h * S_LEN + s) * 128;
    dp[j]      = f2bf_bits(x0 * c0 - x1 * s0);   // out[:h] = x[:h]*cos - x[h:]*sin
    dp[j + 64] = f2bf_bits(x1 * c1 + x0 * s1);   // out[h:] = x[h:]*cos + x[:h]*sin
}

// V relayout (transpose): src [S][NKV*128] -> dst [NKV][128][S]
__global__ void v_transpose_k(const u16* __restrict__ src, u16* __restrict__ dst) {
    int tid = blockIdx.x * blockDim.x + threadIdx.x;
    if (tid >= S_LEN * (NKV * HDIM)) return;
    int c = tid & 511;
    int s = tid >> 9;
    dst[(size_t)c * S_LEN + s] = src[(size_t)s * 512 + c];
}

// ---------------------------------------------------------------------------
// WMMA GEMM: C[M][N] = A[M][K] (bf16 row-major) x BT[N][K] (bf16, B transposed)
// Block = 4 waves; each wave computes a 32x32 tile (2x2 WMMA accumulators),
// software-pipelined: fragments for kc+32 load while kc's WMMAs execute.
// Weights are L2-resident (192MB L2 >> 20MB of bf16 weights) -> direct loads;
// A rows are streamed with global_prefetch.
// ---------------------------------------------------------------------------
template <bool F32OUT>
__global__ __launch_bounds__(128)
void gemm_wmma_k(const u16* __restrict__ A, const u16* __restrict__ BT,
                 void* __restrict__ Cv, int M, int N, int K) {
    const int lane = threadIdx.x;
    const int g    = lane >> 4;       // lane group (K-half selector)
    const int l16  = lane & 15;
    const int wid  = threadIdx.y;
    const int mbase = blockIdx.y * 64 + (wid >> 1) * 32;
    const int nbase = blockIdx.x * 64 + (wid & 1) * 32;

    v8f acc[2][2] = {};
    const u16* ar0 = A  + (size_t)(mbase + l16)      * K;
    const u16* ar1 = A  + (size_t)(mbase + 16 + l16) * K;
    const u16* br0 = BT + (size_t)(nbase + l16)      * K;
    const u16* br1 = BT + (size_t)(nbase + 16 + l16) * K;

    // pipeline prologue
    v16bf a0 = ld_frag_row(ar0, g);
    v16bf a1 = ld_frag_row(ar1, g);
    v16bf b0 = ld_frag_col(br0, g);
    v16bf b1 = ld_frag_col(br1, g);

    for (int kc = 0; kc < K; kc += 32) {
        const int kn = kc + 32;
        v16bf na0 = a0, na1 = a1, nb0 = b0, nb1 = b1;
        if (kn < K) {                              // issue next-stage loads
            if (kn + 256 < K) {                    // + prefetch streamed A
                __builtin_prefetch(ar0 + kn + 256, 0, 0);
                __builtin_prefetch(ar1 + kn + 256, 0, 0);
            }
            na0 = ld_frag_row(ar0 + kn, g);
            na1 = ld_frag_row(ar1 + kn, g);
            nb0 = ld_frag_col(br0 + kn, g);
            nb1 = ld_frag_col(br1 + kn, g);
        }
        acc[0][0] = wmma_bf16(a0, b0, acc[0][0]);
        acc[0][1] = wmma_bf16(a0, b1, acc[0][1]);
        acc[1][0] = wmma_bf16(a1, b0, acc[1][0]);
        acc[1][1] = wmma_bf16(a1, b1, acc[1][1]);
        a0 = na0; a1 = na1; b0 = nb0; b1 = nb1;
    }
    // C layout: VGPR r holds row (r + 8*g), lane%16 holds col.
    #pragma unroll
    for (int i = 0; i < 2; ++i)
        #pragma unroll
        for (int j = 0; j < 2; ++j)
            #pragma unroll
            for (int r = 0; r < 8; ++r) {
                int row = mbase + i * 16 + r + 8 * g;
                int col = nbase + j * 16 + l16;
                if (F32OUT)
                    ((float*)Cv)[(size_t)row * N + col] = acc[i][j][r];
                else
                    ((u16*)Cv)[(size_t)row * N + col] = f2bf_bits(acc[i][j][r]);
            }
}

// ---------------------------------------------------------------------------
// Flash attention, transposed-score formulation: one wave per 16-row Q tile.
//   S' = K_tile x Q^T  =>  C-tile: lane = q column, VGPR element = k row.
//   - softmax stats are per-lane scalars (one shfl_xor(16) merges lane halves)
//   - exp() results are ALREADY in the A-fragment layout for the PV WMMA
//     (row q = lane%16, K offsets {8g+r} / {16+8g+r}) -> no LDS, no barriers.
//   - K frags load as one batch before the score WMMAs; V frags are issued
//     before the softmax math so exp/shuffle work hides their latency.
// Q: [NH][S][128]  K: [NKV][S][128]  Vt: [NKV][128][S]  -> Abf: [S][2048] bf16
// ---------------------------------------------------------------------------
__global__ __launch_bounds__(32)
void attn_wmma_k(const u16* __restrict__ Qbf, const u16* __restrict__ Kbf,
                 const u16* __restrict__ Vt, u16* __restrict__ Abf) {
    const int lane = threadIdx.x;
    const int g    = lane >> 4;
    const int l16  = lane & 15;
    const int h    = blockIdx.y;
    const int kh   = h >> 2;
    const int q0   = blockIdx.x * 16;

    // Q as B-fragments (Q^T: K-dim = head_dim, N = q cols), loaded once
    v16bf qb[4];
    const u16* qrow = Qbf + ((size_t)h * S_LEN + q0 + l16) * HDIM;
    #pragma unroll
    for (int dc = 0; dc < 4; ++dc) qb[dc] = ld_frag_col(qrow + dc * 32, g);

    v8f acc[8] = {};
    float m = -1e30f, lsum = 0.0f;
    const float scale = 0.08838834764831845f;   // 1/sqrt(128)
    const int qcol = q0 + l16;
    const int kend = q0 + 16;

    for (int kc = 0; kc < kend; kc += 32) {
        // ---- batch-load all K fragments for this chunk (one drain point) ----
        const u16* krow0 = Kbf + ((size_t)kh * S_LEN + kc + l16) * HDIM;
        const u16* krow1 = krow0 + 16 * HDIM;
        v16bf ka[8];
        #pragma unroll
        for (int dc = 0; dc < 4; ++dc) {
            ka[dc]     = ld_frag_row(krow0 + dc * 32, g);
            ka[4 + dc] = ld_frag_row(krow1 + dc * 32, g);
        }
        // ---- scores (transposed): two 16x16 tiles, k rows [kc, kc+32) ----
        v8f sc0 = {}, sc1 = {};
        #pragma unroll
        for (int dc = 0; dc < 4; ++dc) {
            sc0 = wmma_bf16(ka[dc],     qb[dc], sc0);
            sc1 = wmma_bf16(ka[4 + dc], qb[dc], sc1);
        }
        // ---- issue V fragment loads; softmax math below hides the latency --
        v16bf vf[8];
        #pragma unroll
        for (int nt = 0; nt < 8; ++nt) {
            const u16* vp = Vt + ((size_t)kh * HDIM + nt * 16 + l16) * S_LEN + kc;
            vf[nt] = ld_frag_col(vp, g);
        }
        // ---- scale + causal mask; element r -> k = kc + {0,16} + r + 8g ----
        float mx = -1e30f;
        #pragma unroll
        for (int r = 0; r < 8; ++r) {
            float a = ((kc + r + 8 * g)      <= qcol) ? sc0[r] * scale : -1e30f;
            float b = ((kc + 16 + r + 8 * g) <= qcol) ? sc1[r] * scale : -1e30f;
            sc0[r] = a;
            sc1[r] = b;
            mx = fmaxf(mx, fmaxf(a, b));
        }
        mx = fmaxf(mx, __shfl_xor(mx, 16));         // merge lane halves
        const float mnew = fmaxf(m, mx);
        const float resc = __expf(m - mnew);
        m = mnew;

        // ---- exp + row sum; pack P directly into the PV A-fragment ----
        v8bf pk0 = {}, pk1 = {};
        float rs = 0.0f;
        #pragma unroll
        for (int r = 0; r < 8; ++r) {
            float p0 = __expf(sc0[r] - mnew);
            float p1 = __expf(sc1[r] - mnew);
            rs += p0 + p1;
            pk0[r] = f2bf(p0);
            pk1[r] = f2bf(p1);
        }
        rs += __shfl_xor(rs, 16);
        lsum = lsum * resc + rs;
        v16bf pf = __builtin_shufflevector(pk0, pk1,
                                           0,1,2,3,4,5,6,7,8,9,10,11,12,13,14,15);

        // ---- rescale accumulators (stats layout -> acc layout: 8 shfl) ----
        float rb[8];
        #pragma unroll
        for (int r = 0; r < 8; ++r) rb[r] = __shfl(resc, r + 8 * g);
        #pragma unroll
        for (int nt = 0; nt < 8; ++nt)
            #pragma unroll
            for (int r = 0; r < 8; ++r) acc[nt][r] *= rb[r];

        // ---- PV: P(16x32) x V(32x128); Vt gives contiguous B fragments ----
        #pragma unroll
        for (int nt = 0; nt < 8; ++nt)
            acc[nt] = wmma_bf16(pf, vf[nt], acc[nt]);
    }

    // ---- normalize + store to [S][2048] for the output projection ----
    float lb[8];
    #pragma unroll
    for (int r = 0; r < 8; ++r) lb[r] = __shfl(lsum, r + 8 * g);
    #pragma unroll
    for (int nt = 0; nt < 8; ++nt)
        #pragma unroll
        for (int r = 0; r < 8; ++r) {
            const int row = q0 + r + 8 * g;
            float o = acc[nt][r] / lb[r];
            Abf[(size_t)row * 2048 + h * 128 + nt * 16 + l16] = f2bf_bits(o);
        }
}

// ---------------------------------------------------------------------------
// Host-side launcher
// ---------------------------------------------------------------------------
extern "C" void kernel_launch(void* const* d_in, const int* in_sizes, int n_in,
                              void* d_out, int out_size, void* d_ws, size_t ws_size,
                              hipStream_t stream) {
    const float* hidden = (const float*)d_in[0];   // [1][4096][2048]
    const float* cosT   = (const float*)d_in[1];   // [4096][128]
    const float* sinT   = (const float*)d_in[2];   // [4096][128]
    const float* Wq     = (const float*)d_in[3];   // [2048][2048]
    const float* Wk     = (const float*)d_in[4];   // [2048][512]
    const float* Wv     = (const float*)d_in[5];   // [2048][512]
    const float* Wo     = (const float*)d_in[6];   // [2048][2048]
    float* out = (float*)d_out;                    // [4096][2048]

    const size_t MB = 1024ull * 1024ull;
    char* ws = (char*)d_ws;
    u16* Hbf  = (u16*)(ws + 0);        // 16MB  [S][2048] bf16
    u16* WqT  = (u16*)(ws + 16 * MB);  //  8MB  [2048][2048]
    u16* WkT  = (u16*)(ws + 24 * MB);  //  2MB  [512][2048]
    u16* WvT  = (u16*)(ws + 26 * MB);  //  2MB  [512][2048]
    u16* WoT  = (u16*)(ws + 28 * MB);  //  8MB  [2048][2048]
    u16* Qtmp = (u16*)(ws + 36 * MB);  // 16MB  [S][2048]
    u16* Ktmp = (u16*)(ws + 52 * MB);  //  4MB  [S][512]
    u16* Vtmp = (u16*)(ws + 56 * MB);  //  4MB  [S][512]
    u16* Qbf  = (u16*)(ws + 60 * MB);  // 16MB  [16][S][128]
    u16* Kbf  = (u16*)(ws + 76 * MB);  //  4MB  [4][S][128]
    u16* Vt   = (u16*)(ws + 80 * MB);  //  4MB  [4][128][S]
    u16* Abf  = Qtmp;                  // reuse Qtmp region after RoPE

    const int TB = 256;
    // 1) precision conversion / weight transposes
    f32_to_bf16_k<<<(S_LEN * DMODEL + TB - 1) / TB, TB, 0, stream>>>(hidden, Hbf, S_LEN * DMODEL);
    f32_transpose_bf16_k<<<(DMODEL * 2048 + TB - 1) / TB, TB, 0, stream>>>(Wq, WqT, DMODEL, 2048);
    f32_transpose_bf16_k<<<(DMODEL * 512  + TB - 1) / TB, TB, 0, stream>>>(Wk, WkT, DMODEL, 512);
    f32_transpose_bf16_k<<<(DMODEL * 512  + TB - 1) / TB, TB, 0, stream>>>(Wv, WvT, DMODEL, 512);
    f32_transpose_bf16_k<<<(DMODEL * 2048 + TB - 1) / TB, TB, 0, stream>>>(Wo, WoT, DMODEL, 2048);

    // 2) QKV projections (WMMA)
    dim3 blk(32, 4);
    gemm_wmma_k<false><<<dim3(2048 / 64, S_LEN / 64), blk, 0, stream>>>(Hbf, WqT, Qtmp, S_LEN, 2048, DMODEL);
    gemm_wmma_k<false><<<dim3(512  / 64, S_LEN / 64), blk, 0, stream>>>(Hbf, WkT, Ktmp, S_LEN, 512,  DMODEL);
    gemm_wmma_k<false><<<dim3(512  / 64, S_LEN / 64), blk, 0, stream>>>(Hbf, WvT, Vtmp, S_LEN, 512,  DMODEL);

    // 3) RoPE + relayout
    rope_relayout_k<<<(S_LEN * NHEADS * 64 + TB - 1) / TB, TB, 0, stream>>>(Qtmp, cosT, sinT, Qbf, NHEADS);
    rope_relayout_k<<<(S_LEN * NKV    * 64 + TB - 1) / TB, TB, 0, stream>>>(Ktmp, cosT, sinT, Kbf, NKV);
    v_transpose_k<<<(S_LEN * NKV * HDIM + TB - 1) / TB, TB, 0, stream>>>(Vtmp, Vt);

    // 4) flash attention (WMMA scores + PV, no LDS staging)
    attn_wmma_k<<<dim3(S_LEN / 16, NHEADS), dim3(32), 0, stream>>>(Qbf, Kbf, Vt, Abf);

    // 5) output projection (WMMA, f32 store)
    gemm_wmma_k<true><<<dim3(2048 / 64, S_LEN / 64), blk, 0, stream>>>(Abf, WoT, out, S_LEN, 2048, DMODEL);

    (void)in_sizes; (void)n_in; (void)out_size; (void)ws_size;
}
